// MultiModalPiMoE_2078764171733
// MI455X (gfx1250) — matrix operand order repaired and checked by reading, not compiled, not run on hardware
//
#include <hip/hip_runtime.h>
#include <math.h>

// ============================================================================
// MultiModal forward for MI455X (gfx1250, wave32, WMMA).
// All dense matmuls lower to v_wmma_f32_16x16x32_f16 (f32 accumulate).
// GEMM inner loop is branch-free and templated on the B-operand layout:
// row indices are clamped (out-of-range rows compute garbage that the store
// guards discard) and K is required to be a multiple of 64 (true for every
// matmul in this model), so the loop issues two WMMAs per iteration fed by
// clauses of b128 loads.
// ============================================================================

typedef __attribute__((ext_vector_type(16))) _Float16 v16h;
typedef __attribute__((ext_vector_type(8)))  float    v8f;

#define ACT_NONE 0
#define ACT_RELU 1

// ---- fragment loaders ------------------------------------------------------
// A fragment (16x32 f16): lane half 0 -> K {kc..kc+15}
//                         lane half 1 -> K {kc+16..kc+31}
__device__ __forceinline__ v16h load_a_frag(const float* __restrict__ Ap,
                                            int kc, int half)
{
    const float4* p0 = (const float4*)(Ap + kc + half * 16);
    const float4* p1 = (const float4*)(Ap + kc + half * 16 + 8);
    float4 a0 = p0[0], a1 = p0[1];
    float4 a2 = p1[0], a3 = p1[1];
    v16h f;
    f[0]  = (_Float16)a0.x; f[1]  = (_Float16)a0.y;
    f[2]  = (_Float16)a0.z; f[3]  = (_Float16)a0.w;
    f[4]  = (_Float16)a1.x; f[5]  = (_Float16)a1.y;
    f[6]  = (_Float16)a1.z; f[7]  = (_Float16)a1.w;
    f[8]  = (_Float16)a2.x; f[9]  = (_Float16)a2.y;
    f[10] = (_Float16)a2.z; f[11] = (_Float16)a2.w;
    f[12] = (_Float16)a3.x; f[13] = (_Float16)a3.y;
    f[14] = (_Float16)a3.z; f[15] = (_Float16)a3.w;
    return f;
}

// B fragment (32x16 f16), W row-major (N x K): contiguous 64B run per lane.
__device__ __forceinline__ v16h load_b_frag_t(const float* __restrict__ Wp, int kb)
{
    const float4* q = (const float4*)(Wp + kb);
    float4 b0 = q[0], b1 = q[1], b2 = q[2], b3 = q[3];
    v16h f;
    f[0]  = (_Float16)b0.x; f[1]  = (_Float16)b0.y;
    f[2]  = (_Float16)b0.z; f[3]  = (_Float16)b0.w;
    f[4]  = (_Float16)b1.x; f[5]  = (_Float16)b1.y;
    f[6]  = (_Float16)b1.z; f[7]  = (_Float16)b1.w;
    f[8]  = (_Float16)b2.x; f[9]  = (_Float16)b2.y;
    f[10] = (_Float16)b2.z; f[11] = (_Float16)b2.w;
    f[12] = (_Float16)b3.x; f[13] = (_Float16)b3.y;
    f[14] = (_Float16)b3.z; f[15] = (_Float16)b3.w;
    return f;
}

// B fragment, W row-major (K x N): strided column gather.
__device__ __forceinline__ v16h load_b_frag_n(const float* __restrict__ W,
                                              int ldw, int kb, int wrow)
{
    v16h f;
#pragma unroll
    for (int j = 0; j < 16; ++j)
        f[j] = (_Float16)W[(size_t)(kb + j) * ldw + wrow];
    return f;
}

// ---------------------------------------------------------------------------
// Generic batched WMMA GEMM:
//   D = act( alpha * A(MxK) * op(W) + bias + res )
//   TRANSB=1: W is (N x K) row-major (use W^T)    -> linear projections
//   TRANSB=0: W is (K x N) row-major              -> P @ V
// Batch index z decomposes as (zo, zi) with zi in [0, batchInner) so that
// attention heads ((b,h) pairs) batch in one launch.
// One wave computes one 16x16 tile; block = 8 waves = 32x64 output region.
// Requires K % 64 == 0 (true for every matmul in this model).
// ---------------------------------------------------------------------------
template <int TRANSB>
__global__ void __launch_bounds__(256)
wmma_gemm_kernel(const float* __restrict__ A, const float* __restrict__ W,
                 const float* __restrict__ bias, const float* __restrict__ res,
                 float* __restrict__ D,
                 int M, int N, int K, int lda, int ldw, int ldres, int ldd,
                 float alpha, int act, int batchInner,
                 long long sAo, long long sAi, long long sWo, long long sWi,
                 long long sDo, long long sDi)
{
    int z  = blockIdx.z;
    int zo = z / batchInner;
    int zi = z - zo * batchInner;
    A += (size_t)zo * sAo + (size_t)zi * sAi;
    W += (size_t)zo * sWo + (size_t)zi * sWi;
    D += (size_t)zo * sDo + (size_t)zi * sDi;

    int tid  = threadIdx.x;
    int wave = tid >> 5;
    int lane = tid & 31;
    int half = lane >> 4;   // which 16-lane half of the wave
    int r    = lane & 15;   // row-in-tile for A, col-in-tile for B

    int m0 = (blockIdx.y * 2 + (wave >> 2)) * 16;
    int n0 = (blockIdx.x * 4 + (wave & 3)) * 16;
    if (m0 >= M || n0 >= N) return;   // uniform per wave -> WMMA EXEC stays all-1

    // Clamp rows instead of zero-filling: an out-of-range A row only affects
    // output rows >= M and an out-of-range W row only affects output cols
    // >= N, both of which are discarded by the store guards below.
    int arow = min(m0 + r, M - 1);
    int wrow = min(n0 + r, N - 1);
    const float* Ap = A + (size_t)arow * lda;
    const float* Wp = W + (size_t)wrow * ldw;   // used when TRANSB

    v8f acc = {};

    for (int kc = 0; kc < K; kc += 64) {   // requires K % 64 == 0
        v16h a0 = load_a_frag(Ap, kc, half);
        v16h a1 = load_a_frag(Ap, kc + 32, half);
        v16h b0, b1;
        if (TRANSB) {
            b0 = load_b_frag_t(Wp, kc + half * 16);
            b1 = load_b_frag_t(Wp, kc + 32 + half * 16);
        } else {
            b0 = load_b_frag_n(W, ldw, kc + half * 16, wrow);
            b1 = load_b_frag_n(W, ldw, kc + 32 + half * 16, wrow);
        }
        acc = __builtin_amdgcn_wmma_f32_16x16x32_f16(
            false, a0, false, b0, (short)0, acc, false, false);
        acc = __builtin_amdgcn_wmma_f32_16x16x32_f16(
            false, a1, false, b1, (short)0, acc, false, false);
    }

    // C/D layout: VGPR i holds M = m0 + i + 8*half, N = n0 + r
    int n = n0 + r;
    if (n < N) {
        float bv = bias ? bias[n] : 0.0f;
        if (m0 + 16 <= M) {
#pragma unroll
            for (int i = 0; i < 8; ++i) {
                int m = m0 + half * 8 + i;
                float v = acc[i] * alpha + bv;
                if (res) v += res[(size_t)m * ldres + n];
                if (act == ACT_RELU) v = fmaxf(v, 0.0f);
                D[(size_t)m * ldd + n] = v;
            }
        } else {
#pragma unroll
            for (int i = 0; i < 8; ++i) {
                int m = m0 + half * 8 + i;
                if (m < M) {
                    float v = acc[i] * alpha + bv;
                    if (res) v += res[(size_t)m * ldres + n];
                    if (act == ACT_RELU) v = fmaxf(v, 0.0f);
                    D[(size_t)m * ldd + n] = v;
                }
            }
        }
    }
}

static void gemm(hipStream_t st, int M, int N, int K,
                 const float* A, int lda,
                 const float* W, int ldw, int transB,
                 const float* bias, const float* res, int ldres,
                 float* D, int ldd, float alpha, int act,
                 int batch = 1, int batchInner = 1,
                 long long sAo = 0, long long sAi = 0,
                 long long sWo = 0, long long sWi = 0,
                 long long sDo = 0, long long sDi = 0)
{
    dim3 grid((N + 63) / 64, (M + 31) / 32, batch);
    if (transB)
        wmma_gemm_kernel<1><<<grid, 256, 0, st>>>(A, W, bias, res, D, M, N, K,
                                                  lda, ldw, ldres, ldd, alpha, act,
                                                  batchInner,
                                                  sAo, sAi, sWo, sWi, sDo, sDi);
    else
        wmma_gemm_kernel<0><<<grid, 256, 0, st>>>(A, W, bias, res, D, M, N, K,
                                                  lda, ldw, ldres, ldd, alpha, act,
                                                  batchInner,
                                                  sAo, sAi, sWo, sWi, sDo, sDi);
}

// ---------------------------------------------------------------------------
// Elementwise / reduction kernels (bandwidth-bound; plain VALU)
// ---------------------------------------------------------------------------
__global__ void embed_kernel(const int* __restrict__ tok,
                             const float* __restrict__ emb,
                             const float* __restrict__ pos,
                             float* __restrict__ x, int B, int L, int Hd)
{
    int idx = blockIdx.x * blockDim.x + threadIdx.x;
    if (idx >= B * L * Hd) return;
    int h = idx % Hd; int t = idx / Hd; int l = t % L; int b = t / L;
    x[idx] = emb[(size_t)tok[b * L + l] * Hd + h] + pos[(size_t)l * Hd + h];
}

__global__ void softmax_rows_kernel(float* __restrict__ X, int rows, int cols)
{
    int row = blockIdx.x;
    if (row >= rows) return;
    float* x = X + (size_t)row * cols;
    __shared__ float red[256];
    float mx = -1e30f;
    for (int c = threadIdx.x; c < cols; c += 256) mx = fmaxf(mx, x[c]);
    red[threadIdx.x] = mx; __syncthreads();
    for (int s = 128; s > 0; s >>= 1) {
        if (threadIdx.x < s) red[threadIdx.x] = fmaxf(red[threadIdx.x], red[threadIdx.x + s]);
        __syncthreads();
    }
    mx = red[0]; __syncthreads();
    float sum = 0.0f;
    for (int c = threadIdx.x; c < cols; c += 256) {
        float e = __expf(x[c] - mx); x[c] = e; sum += e;
    }
    red[threadIdx.x] = sum; __syncthreads();
    for (int s = 128; s > 0; s >>= 1) {
        if (threadIdx.x < s) red[threadIdx.x] += red[threadIdx.x + s];
        __syncthreads();
    }
    float inv = 1.0f / red[0];
    for (int c = threadIdx.x; c < cols; c += 256) x[c] *= inv;
}

// Y[row] = ((X[row]-mean)/sqrt(var+eps)*g + b) * (*scale)
__global__ void layernorm_kernel(const float* __restrict__ X,
                                 const float* __restrict__ g,
                                 const float* __restrict__ bta,
                                 float* __restrict__ Y,
                                 int rows, int cols, int ldx, int ldy,
                                 const float* __restrict__ scale_ptr)
{
    int row = blockIdx.x;
    if (row >= rows) return;
    const float* x = X + (size_t)row * ldx;
    float*       y = Y + (size_t)row * ldy;
    __shared__ float red[256];
    float s = 0.0f;
    for (int c = threadIdx.x; c < cols; c += 256) s += x[c];
    red[threadIdx.x] = s; __syncthreads();
    for (int st = 128; st > 0; st >>= 1) {
        if (threadIdx.x < st) red[threadIdx.x] += red[threadIdx.x + st];
        __syncthreads();
    }
    float mean = red[0] / cols; __syncthreads();
    float v = 0.0f;
    for (int c = threadIdx.x; c < cols; c += 256) {
        float d = x[c] - mean; v += d * d;
    }
    red[threadIdx.x] = v; __syncthreads();
    for (int st = 128; st > 0; st >>= 1) {
        if (threadIdx.x < st) red[threadIdx.x] += red[threadIdx.x + st];
        __syncthreads();
    }
    float inv = rsqrtf(red[0] / cols + 1e-5f);
    float sc  = scale_ptr ? *scale_ptr : 1.0f;
    for (int c = threadIdx.x; c < cols; c += 256)
        y[c] = ((x[c] - mean) * inv * g[c] + bta[c]) * sc;
}

// out[b,c] = mean over T of in[b,t,c]
__global__ void mean_axis1_kernel(const float* __restrict__ X, float* __restrict__ Y,
                                  int B, int T, int C)
{
    int i = blockIdx.x * blockDim.x + threadIdx.x;
    if (i >= B * C) return;
    int b = i / C, c = i % C;
    const float* x = X + (size_t)b * T * C + c;
    float s = 0.0f;
    for (int t = 0; t < T; ++t) s += x[(size_t)t * C];
    Y[i] = s / (float)T;
}

__global__ void vec_add_kernel(const float* __restrict__ a, const float* __restrict__ b,
                               float* __restrict__ y, int n)
{
    int i = blockIdx.x * blockDim.x + threadIdx.x;
    if (i < n) y[i] = a[i] + b[i];
}

// ---------------------------------------------------------------------------
// Conv stems (direct; bn scale = g / sqrt(1+eps))
// ---------------------------------------------------------------------------
#define BN_RS 0.9999950000374997f

__global__ void conv2d_bn_relu_kernel(const float* __restrict__ X,
                                      const float* __restrict__ Wt,
                                      const float* __restrict__ bias,
                                      const float* __restrict__ bng,
                                      const float* __restrict__ bnb,
                                      float* __restrict__ Y,
                                      int Bn, int Cin, int Hin, int Win,
                                      int Cout, int Kk, int stride, int pad,
                                      int Hout, int Wout)
{
    int idx = blockIdx.x * blockDim.x + threadIdx.x;
    int total = Bn * Cout * Hout * Wout;
    if (idx >= total) return;
    int ox = idx % Wout; int t = idx / Wout;
    int oy = t % Hout;   t /= Hout;
    int co = t % Cout;   int b = t / Cout;
    float acc = bias[co];
    const float* wbase = Wt + (size_t)co * Cin * Kk * Kk;
    for (int ci = 0; ci < Cin; ++ci) {
        const float* xb = X + ((size_t)(b * Cin + ci) * Hin) * Win;
        const float* wb = wbase + (size_t)ci * Kk * Kk;
        for (int ky = 0; ky < Kk; ++ky) {
            int iy = oy * stride - pad + ky;
            if (iy < 0 || iy >= Hin) continue;
            for (int kx = 0; kx < Kk; ++kx) {
                int ix = ox * stride - pad + kx;
                if (ix < 0 || ix >= Win) continue;
                acc += xb[(size_t)iy * Win + ix] * wb[ky * Kk + kx];
            }
        }
    }
    acc = acc * bng[co] * BN_RS + bnb[co];
    Y[idx] = fmaxf(acc, 0.0f);
}

__global__ void conv1d_bn_relu_kernel(const float* __restrict__ X,
                                      const float* __restrict__ Wt,
                                      const float* __restrict__ bias,
                                      const float* __restrict__ bng,
                                      const float* __restrict__ bnb,
                                      float* __restrict__ Y,
                                      int Bn, int Cin, int T, int Cout)
{
    int idx = blockIdx.x * blockDim.x + threadIdx.x;
    int total = Bn * Cout * T;
    if (idx >= total) return;
    int t  = idx % T;  int r = idx / T;
    int co = r % Cout; int b = r / Cout;
    float acc = bias[co];
    const float* wbase = Wt + (size_t)co * Cin * 3;
    for (int ci = 0; ci < Cin; ++ci) {
        const float* xb = X + ((size_t)(b * Cin + ci)) * T;
        const float* wb = wbase + ci * 3;
        for (int k = 0; k < 3; ++k) {
            int it = t - 1 + k;
            if (it < 0 || it >= T) continue;
            acc += xb[it] * wb[k];
        }
    }
    acc = acc * bng[co] * BN_RS + bnb[co];
    Y[idx] = fmaxf(acc, 0.0f);
}

__global__ void avgpool2d_kernel(const float* __restrict__ X, float* __restrict__ Y,
                                 int BC, int HW)
{
    int i = blockIdx.x * blockDim.x + threadIdx.x;
    if (i >= BC) return;
    const float* x = X + (size_t)i * HW;
    float s = 0.0f;
    for (int j = 0; j < HW; ++j) s += x[j];
    Y[i] = s / (float)HW;
}

__global__ void transpose_bct_btc_kernel(const float* __restrict__ X,
                                         float* __restrict__ Y,
                                         int B, int C, int T)
{
    int i = blockIdx.x * blockDim.x + threadIdx.x;
    if (i >= B * C * T) return;
    int t = i % T; int r = i / T; int c = r % C; int b = r / C;
    Y[((size_t)b * T + t) * C + c] = X[i];
}

// ---------------------------------------------------------------------------
// LSTM pointwise cell: g = pre_t + h@Whh^T (tmp); update h,c; write h to out.
// ---------------------------------------------------------------------------
__global__ void lstm_cell_kernel(const float* __restrict__ pre_t, long long preBstride,
                                 const float* __restrict__ tmp,
                                 float* __restrict__ h, float* __restrict__ c,
                                 float* __restrict__ out_t, long long outBstride,
                                 int B, int Hh)
{
    int idx = blockIdx.x * blockDim.x + threadIdx.x;
    if (idx >= B * Hh) return;
    int b = idx / Hh, j = idx % Hh;
    const float* g = pre_t + (size_t)b * preBstride;
    const float* u = tmp + (size_t)b * 4 * Hh;
    float gi = g[j]          + u[j];
    float gf = g[Hh + j]     + u[Hh + j];
    float gg = g[2 * Hh + j] + u[2 * Hh + j];
    float go = g[3 * Hh + j] + u[3 * Hh + j];
    float si = 1.0f / (1.0f + __expf(-gi));
    float sf = 1.0f / (1.0f + __expf(-gf));
    float so = 1.0f / (1.0f + __expf(-go));
    float cc = sf * c[idx] + si * tanhf(gg);
    float hh = so * tanhf(cc);
    c[idx] = cc;
    h[idx] = hh;
    out_t[(size_t)b * outBstride + j] = hh;
}

// ---------------------------------------------------------------------------
// Tiny cross attention (q_len=1, kv_len small). One block per (b,h), 64 thr.
// ---------------------------------------------------------------------------
__global__ void tiny_attn_kernel(const float* __restrict__ Q,
                                 const float* __restrict__ K,
                                 const float* __restrict__ V,
                                 float* __restrict__ O,
                                 int B, int NHc, int hd, int kvlen)
{
    int bh = blockIdx.x;
    int b = bh / NHc, h = bh % NHc;
    int d = threadIdx.x;
    int E = NHc * hd;
    const float* q = Q + (size_t)b * E + h * hd;
    const float* k = K + (size_t)b * kvlen * E + h * hd;
    const float* v = V + (size_t)b * kvlen * E + h * hd;
    __shared__ float red[64];
    __shared__ float s[8];
    float qd = q[d];
    for (int kk = 0; kk < kvlen; ++kk) {
        red[d] = qd * k[(size_t)kk * E + d];
        __syncthreads();
        for (int st = 32; st > 0; st >>= 1) {
            if (d < st) red[d] += red[d + st];
            __syncthreads();
        }
        if (d == 0) s[kk] = red[0] / sqrtf((float)hd);
        __syncthreads();
    }
    if (d == 0) {
        float mx = s[0];
        for (int kk = 1; kk < kvlen; ++kk) mx = fmaxf(mx, s[kk]);
        float sum = 0.0f;
        for (int kk = 0; kk < kvlen; ++kk) { s[kk] = __expf(s[kk] - mx); sum += s[kk]; }
        for (int kk = 0; kk < kvlen; ++kk) s[kk] /= sum;
    }
    __syncthreads();
    float o = 0.0f;
    for (int kk = 0; kk < kvlen; ++kk) o += s[kk] * v[(size_t)kk * E + d];
    O[(size_t)b * E + h * hd + d] = o;
}

__global__ void pack2_kernel(const float* __restrict__ A, const float* __restrict__ Bp,
                             float* __restrict__ Y, int B, int C)
{
    int i = blockIdx.x * blockDim.x + threadIdx.x;
    if (i >= B * C) return;
    int b = i / C, c = i % C;
    Y[((size_t)b * 2) * C + c]     = A[i];
    Y[((size_t)b * 2 + 1) * C + c] = Bp[i];
}

// ---------------------------------------------------------------------------
// MoE gate + combine
// ---------------------------------------------------------------------------
__global__ void moe_gate_kernel(const float* __restrict__ logits,
                                float* __restrict__ w, int B, int E)
{
    int b = blockIdx.x * blockDim.x + threadIdx.x;
    if (b >= B) return;
    const float* l = logits + (size_t)b * E;
    int i0 = 0, i1 = 0; float v0 = -1e30f, v1 = -1e30f;
    for (int e = 0; e < E; ++e) {
        float v = l[e];
        if (v > v0)      { v1 = v0; i1 = i0; v0 = v; i0 = e; }
        else if (v > v1) { v1 = v;  i1 = e; }
    }
    float e0 = __expf(v0 - v0), e1 = __expf(v1 - v0);
    float s = e0 + e1;
    float* wr = w + (size_t)b * E;
    for (int e = 0; e < E; ++e) wr[e] = 0.0f;
    wr[i0] = e0 / s; wr[i1] = e1 / s;
}

__global__ void moe_combine_kernel(const float* __restrict__ w,
                                   const float* __restrict__ ye,
                                   float* __restrict__ out, int B, int E, int Dd)
{
    int idx = blockIdx.x * blockDim.x + threadIdx.x;
    if (idx >= B * Dd) return;
    int b = idx / Dd, d = idx % Dd;
    float acc = 0.0f;
    for (int e = 0; e < E; ++e)
        acc += w[b * E + e] * ye[((size_t)e * B + b) * Dd + d];
    out[idx] = acc;
}

// ===========================================================================
// Host orchestration
// ===========================================================================
extern "C" void kernel_launch(void* const* d_in, const int* in_sizes, int n_in,
                              void* d_out, int out_size, void* d_ws, size_t ws_size,
                              hipStream_t stream)
{
    (void)in_sizes; (void)n_in; (void)out_size; (void)ws_size;

    constexpr int B = 16, L = 256, H = 512, NH = 8, HD = 64, T = 400;
    constexpr int BL = B * L;

#define PF(i) ((const float*)d_in[(i)])
    const int* tokens = (const int*)d_in[0];
    const float* images = PF(1);
    const float* audio  = PF(2);
    // Param flat order (dict insertion order):
    // 3 emb, 4 pos, 5..28 text_layers (12 each), 29/30 t_proj,
    // 31..34 ic_w, 35..38 ic_b, 39..42 ibn_g, 43..46 ibn_b, 47/48 i_proj,
    // 49..51 ac_w, 52..54 ac_b, 55..57 abn_g, 58..60 abn_b,
    // 61..76 lstm (l0f,l0b,l1f,l1b x {wih,whh,bih,bhh}), 77/78 a_proj,
    // 79..108 cm (10 each x 3), 109 fusion_w, 110..115 fn, 116/117 gate,
    // 118..121 experts.

    // ---- workspace bump allocator ----
    size_t off = 0;
    auto alloc = [&](size_t nfloats) {
        float* p = (float*)((char*)d_ws + off);
        off = (off + nfloats * sizeof(float) + 255) & ~(size_t)255;
        return p;
    };
    float* x       = alloc((size_t)BL * H);
    float* qkv     = alloc((size_t)BL * 3 * H);
    float* scores  = alloc((size_t)B * NH * L * L);
    float* attnbuf = alloc((size_t)BL * H);
    float* t1      = alloc((size_t)BL * H);
    float* tmean   = alloc(B * H);
    float* text_emb= alloc(B * H);
    float* y0 = alloc((size_t)B * 64 * 112 * 112);
    float* y1 = alloc((size_t)B * 128 * 56 * 56);
    float* y2 = alloc((size_t)B * 256 * 28 * 28);
    float* y3 = alloc((size_t)B * 512 * 14 * 14);
    float* pooled  = alloc(B * 512);
    float* img_emb = alloc(B * H);
    float* a0 = alloc((size_t)B * 128 * T);
    float* a1 = alloc((size_t)B * 256 * T);
    float* a2 = alloc((size_t)B * 512 * T);
    float* a_btc = alloc((size_t)B * T * 512);
    float* pre   = alloc((size_t)B * T * 2048);
    float* lstm0 = alloc((size_t)B * T * 1024);
    float* lstm1 = alloc((size_t)B * T * 1024);
    float* hbuf  = alloc(B * 512);
    float* cbuf  = alloc(B * 512);
    float* tmp4h = alloc(B * 2048);
    float* bsum  = alloc(2048);
    float* aud_mean = alloc(B * 1024);
    float* aud_emb  = alloc(B * H);
    float* kvbuf = alloc(B * 2 * H);
    float* Qp    = alloc(B * H);
    float* Kp    = alloc(B * 2 * H);
    float* Vp    = alloc(B * 2 * H);
    float* Obuf  = alloc(B * H);
    float* tsm   = alloc(B * H);
    float* xattn = alloc(B * H);
    float* h1c   = alloc(B * 2048);
    float* y2c   = alloc(B * H);
    float* catbuf= alloc(B * 3 * H);
    float* fh    = alloc(B * 1024);
    float* fy    = alloc(B * H);
    float* fusedb= alloc(B * H);
    float* logits= alloc(B * 8);
    float* wgate = alloc(B * 8);
    float* he    = alloc(B * 2048);
    float* ye    = alloc((size_t)8 * B * H);

    // =============================== TEXT ===============================
    {
        int total = BL * H;
        embed_kernel<<<(total + 255) / 256, 256, 0, stream>>>(
            tokens, PF(3), PF(4), x, B, L, H);
    }
    for (int layer = 0; layer < 2; ++layer) {
        int pb = 5 + layer * 12;
        const float *in_w = PF(pb+0), *in_b = PF(pb+1), *out_w = PF(pb+2),
                    *out_b = PF(pb+3), *ln1g = PF(pb+4), *ln1b = PF(pb+5),
                    *ln2g = PF(pb+6), *ln2b = PF(pb+7), *ff1w = PF(pb+8),
                    *ff1b = PF(pb+9), *ff2w = PF(pb+10), *ff2b = PF(pb+11);
        // QKV projection: [BL,1536]
        gemm(stream, BL, 3 * H, H, x, H, in_w, H, 1, in_b, nullptr, 0,
             qkv, 3 * H, 1.0f, ACT_NONE);
        // scores[b,h] = Q Kt / 8, batched over 128 (b,h)
        gemm(stream, L, L, HD,
             qkv, 3 * H, qkv + H, 3 * H, 1, nullptr, nullptr, 0,
             scores, L, 0.125f, ACT_NONE,
             B * NH, NH,
             (long long)L * 3 * H, HD,
             (long long)L * 3 * H, HD,
             (long long)NH * L * L, (long long)L * L);
        softmax_rows_kernel<<<B * NH * L, 256, 0, stream>>>(scores, B * NH * L, L);
        // O[b,h] = P V   (V row-major -> transB = 0)
        gemm(stream, L, HD, L,
             scores, L, qkv + 2 * H, 3 * H, 0, nullptr, nullptr, 0,
             attnbuf, H, 1.0f, ACT_NONE,
             B * NH, NH,
             (long long)NH * L * L, (long long)L * L,
             (long long)L * 3 * H, HD,
             (long long)L * H, HD);
        // out projection + residual, then LN -> x
        gemm(stream, BL, H, H, attnbuf, H, out_w, H, 1, out_b, x, H,
             t1, H, 1.0f, ACT_NONE);
        layernorm_kernel<<<BL, 256, 0, stream>>>(t1, ln1g, ln1b, x, BL, H, H, H, nullptr);
        // FFN (H -> H -> H) + residual, LN -> x
        gemm(stream, BL, H, H, x, H, ff1w, H, 1, ff1b, nullptr, 0,
             attnbuf, H, 1.0f, ACT_RELU);
        gemm(stream, BL, H, H, attnbuf, H, ff2w, H, 1, ff2b, x, H,
             t1, H, 1.0f, ACT_NONE);
        layernorm_kernel<<<BL, 256, 0, stream>>>(t1, ln2g, ln2b, x, BL, H, H, H, nullptr);
    }
    mean_axis1_kernel<<<(B * H + 255) / 256, 256, 0, stream>>>(x, tmean, B, L, H);
    gemm(stream, B, H, H, tmean, H, PF(29), H, 1, PF(30), nullptr, 0,
         text_emb, H, 1.0f, ACT_NONE);

    // =============================== IMAGE ==============================
    {
        struct { const float* in; float* out; int ci, hi, wi, co, k, pad; } cl[4] = {
            { images, y0,   3, 224, 224,  64, 7, 3 },
            { y0,     y1,  64, 112, 112, 128, 3, 1 },
            { y1,     y2, 128,  56,  56, 256, 3, 1 },
            { y2,     y3, 256,  28,  28, 512, 3, 1 },
        };
        for (int i = 0; i < 4; ++i) {
            int ho = (cl[i].hi + 2 * cl[i].pad - cl[i].k) / 2 + 1;
            int wo = (cl[i].wi + 2 * cl[i].pad - cl[i].k) / 2 + 1;
            int total = B * cl[i].co * ho * wo;
            conv2d_bn_relu_kernel<<<(total + 255) / 256, 256, 0, stream>>>(
                cl[i].in, PF(31 + i), PF(35 + i), PF(39 + i), PF(43 + i),
                cl[i].out, B, cl[i].ci, cl[i].hi, cl[i].wi,
                cl[i].co, cl[i].k, 2, cl[i].pad, ho, wo);
        }
        avgpool2d_kernel<<<(B * 512 + 255) / 256, 256, 0, stream>>>(y3, pooled, B * 512, 14 * 14);
        gemm(stream, B, H, 512, pooled, 512, PF(47), 512, 1, PF(48), nullptr, 0,
             img_emb, H, 1.0f, ACT_NONE);
    }

    // =============================== AUDIO ==============================
    {
        struct { const float* in; float* out; int ci, co; } al[3] = {
            { audio, a0,  80, 128 }, { a0, a1, 128, 256 }, { a1, a2, 256, 512 },
        };
        for (int i = 0; i < 3; ++i) {
            int total = B * al[i].co * T;
            conv1d_bn_relu_kernel<<<(total + 255) / 256, 256, 0, stream>>>(
                al[i].in, PF(49 + i), PF(52 + i), PF(55 + i), PF(58 + i),
                al[i].out, B, al[i].ci, T, al[i].co);
        }
        int total = B * 512 * T;
        transpose_bct_btc_kernel<<<(total + 255) / 256, 256, 0, stream>>>(a2, a_btc, B, 512, T);

        // bi-LSTM, 2 layers
        for (int layer = 0; layer < 2; ++layer) {
            int din = layer ? 1024 : 512;
            const float* xin = layer ? lstm0 : a_btc;
            float* xout = layer ? lstm1 : lstm0;
            for (int dir = 0; dir < 2; ++dir) {
                int pb = 61 + layer * 8 + dir * 4;
                const float *wih = PF(pb), *whh = PF(pb + 1),
                            *bih = PF(pb + 2), *bhh = PF(pb + 3);
                vec_add_kernel<<<(2048 + 255) / 256, 256, 0, stream>>>(bih, bhh, bsum, 2048);
                // pre[b*T+t, 2048] = x @ wih^T + (bih+bhh)
                gemm(stream, B * T, 2048, din, xin, din, wih, din, 1, bsum,
                     nullptr, 0, pre, 2048, 1.0f, ACT_NONE);
                hipMemsetAsync(hbuf, 0, B * 512 * sizeof(float), stream);
                hipMemsetAsync(cbuf, 0, B * 512 * sizeof(float), stream);
                for (int s = 0; s < T; ++s) {
                    int t = dir ? (T - 1 - s) : s;
                    gemm(stream, B, 2048, 512, hbuf, 512, whh, 512, 1,
                         nullptr, nullptr, 0, tmp4h, 2048, 1.0f, ACT_NONE);
                    lstm_cell_kernel<<<(B * 512 + 255) / 256, 256, 0, stream>>>(
                        pre + (size_t)t * 2048, (long long)T * 2048,
                        tmp4h, hbuf, cbuf,
                        xout + (size_t)t * 1024 + dir * 512, (long long)T * 1024,
                        B, 512);
                }
            }
        }
        mean_axis1_kernel<<<(B * 1024 + 255) / 256, 256, 0, stream>>>(lstm1, aud_mean, B, T, 1024);
        gemm(stream, B, H, 1024, aud_mean, 1024, PF(77), 1024, 1, PF(78),
             nullptr, 0, aud_emb, H, 1.0f, ACT_NONE);
    }

    // =========================== CROSS-ATTN FUSION ======================
    const float* embs[3] = { text_emb, img_emb, aud_emb };
    for (int i = 0; i < 3; ++i) {
        int pb = 79 + i * 10;
        const float *in_w = PF(pb), *in_b = PF(pb+1), *out_w = PF(pb+2),
                    *out_b = PF(pb+3), *lng = PF(pb+4), *lnb = PF(pb+5),
                    *ff1w = PF(pb+6), *ff1b = PF(pb+7), *ff2w = PF(pb+8),
                    *ff2b = PF(pb+9);
        const float* ka = embs[(i == 0) ? 1 : 0];
        const float* kb = embs[(i == 2) ? 1 : 2];
        pack2_kernel<<<(B * H + 255) / 256, 256, 0, stream>>>(ka, kb, kvbuf, B, H);
        gemm(stream, B, H, H, embs[i], H, in_w, H, 1, in_b, nullptr, 0,
             Qp, H, 1.0f, ACT_NONE);
        gemm(stream, 2 * B, H, H, kvbuf, H, in_w + (size_t)H * H, H, 1,
             in_b + H, nullptr, 0, Kp, H, 1.0f, ACT_NONE);
        gemm(stream, 2 * B, H, H, kvbuf, H, in_w + (size_t)2 * H * H, H, 1,
             in_b + 2 * H, nullptr, 0, Vp, H, 1.0f, ACT_NONE);
        tiny_attn_kernel<<<B * NH, HD, 0, stream>>>(Qp, Kp, Vp, Obuf, B, NH, HD, 2);
        gemm(stream, B, H, H, Obuf, H, out_w, H, 1, out_b, embs[i], H,
             tsm, H, 1.0f, ACT_NONE);
        layernorm_kernel<<<B, 256, 0, stream>>>(tsm, lng, lnb, xattn, B, H, H, H, nullptr);
        gemm(stream, B, 2048, H, xattn, H, ff1w, H, 1, ff1b, nullptr, 0,
             h1c, 2048, 1.0f, ACT_RELU);
        gemm(stream, B, H, 2048, h1c, 2048, ff2w, 2048, 1, ff2b, xattn, H,
             y2c, H, 1.0f, ACT_NONE);
        // LN * fusion_w[i], written directly into cat buffer column slice
        layernorm_kernel<<<B, 256, 0, stream>>>(y2c, lng, lnb,
                                                catbuf + i * H, B, H, H, 3 * H,
                                                PF(109) + i);
    }

    // ============================= FUSION FFN ===========================
    gemm(stream, B, 1024, 3 * H, catbuf, 3 * H, PF(110), 3 * H, 1, PF(111),
         nullptr, 0, fh, 1024, 1.0f, ACT_RELU);
    gemm(stream, B, H, 1024, fh, 1024, PF(112), 1024, 1, PF(113),
         nullptr, 0, fy, H, 1.0f, ACT_NONE);
    layernorm_kernel<<<B, 256, 0, stream>>>(fy, PF(114), PF(115), fusedb, B, H, H, H, nullptr);

    // ================================ MoE ===============================
    gemm(stream, B, 8, H, fusedb, H, PF(116), H, 1, PF(117), nullptr, 0,
         logits, 8, 1.0f, ACT_NONE);
    moe_gate_kernel<<<1, 32, 0, stream>>>(logits, wgate, B, 8);
    for (int e = 0; e < 8; ++e) {
        gemm(stream, B, 2048, H, fusedb, H,
             PF(118) + (size_t)e * 2048 * H, H, 1,
             PF(119) + (size_t)e * 2048, nullptr, 0, he, 2048, 1.0f, ACT_RELU);
        gemm(stream, B, H, 2048, he, 2048,
             PF(120) + (size_t)e * H * 2048, 2048, 1,
             PF(121) + (size_t)e * H, nullptr, 0,
             ye + (size_t)e * B * H, H, 1.0f, ACT_NONE);
    }
    moe_combine_kernel<<<(B * H + 255) / 256, 256, 0, stream>>>(
        wgate, ye, (float*)d_out, B, 8, H);
#undef PF
}
